// PISID_62070867362275
// MI455X (gfx1250) — matrix operand order repaired and checked by reading, not compiled, not run on hardware
//
#include <hip/hip_runtime.h>
#include <hip/hip_bf16.h>
#include <math.h>

// ---------------- problem constants ----------------
#define BATCH   32
#define INLEN   28
#define NNODES  2048
#define EDIM    256
#define NDDIM   256
#define HDIM    512
#define NLAY    3
#define OUTLEN  28
#define MROWS   (BATCH * NNODES)   // 65536

typedef __attribute__((ext_vector_type(16))) __bf16 v16bf;
typedef __attribute__((ext_vector_type(8)))  float  v8f;

union FragU {
    v16bf v;
    unsigned int u[8];
    unsigned short s[16];
};

// CDNA5 async global->LDS DMA (VGLOBAL GLOBAL_LOAD_ASYNC_TO_LDS_B128, ASYNCcnt)
__device__ __forceinline__ void async_ld_b128(unsigned lds_off, const void* gaddr)
{
    asm volatile("global_load_async_to_lds_b128 %0, %1, off"
                 :: "v"(lds_off), "v"(gaddr) : "memory");
}
__device__ __forceinline__ void wait_asynccnt0()
{
    asm volatile("s_wait_asynccnt 0x0" ::: "memory");
}

// ---------------- weight fp32 -> bf16 conversion ----------------
__global__ __launch_bounds__(256)
void k_convert_weights(const float* __restrict__ W1,
                       const float* __restrict__ W2,
                       __bf16* __restrict__ W1b,
                       __bf16* __restrict__ W2b)
{
    int gid = blockIdx.x * 256 + threadIdx.x;   // < NLAY*HDIM*HDIM
    W1b[gid] = (__bf16)W1[gid];
    W2b[gid] = (__bf16)W2[gid];
}

// ---------------- build h0 = [ts | node_emb], fp32 + bf16 ----------------
__global__ __launch_bounds__(256)
void k_build_h0(const float* __restrict__ x_node,   // (B, INLEN, NN, 1)
                const float* __restrict__ node_emb, // (NN, NDDIM)
                const float* __restrict__ W_ts,     // (EDIM, INLEN)
                const float* __restrict__ b_ts,     // (EDIM)
                float*  __restrict__ hf,
                __bf16* __restrict__ hb)
{
    int gid = blockIdx.x * 256 + threadIdx.x;   // < MROWS*HDIM
    int m = gid / HDIM;
    int n = gid - m * HDIM;
    int batch = m >> 11;          // / NNODES
    int node  = m & (NNODES - 1);

    float v;
    if (n < EDIM) {
        float acc = b_ts[n];
        const float* w = W_ts + n * INLEN;
        #pragma unroll
        for (int t = 0; t < INLEN; ++t) {
            // flat[m][t] = x_node[batch, t, node, 0]
            acc += x_node[((size_t)(batch * INLEN + t) << 11) + node] * w[t];
        }
        v = acc;
    } else {
        v = node_emb[(size_t)node * NDDIM + (n - EDIM)];
    }
    hf[gid] = v;
    hb[gid] = (__bf16)v;
}

// ---------------- bf16 WMMA GEMM, fused epilogues ----------------
// C[M,N] = A[M,K] * Bw[N,K]^T   (M=65536, N=K=512)
// mode 0: z = relu(C + bias)            -> zb (bf16)
// mode 1: h = hf + C + bias             -> hf (f32) and hb (bf16)
// Block tile 128(M) x 128(N), 8 waves arranged 4(M) x 2(N).
// Each wave: 32x64 via 2 A-frags x 4 B-frags -> 8 accumulators of
// v_wmma_f32_16x16x32_bf16. Double-buffered LDS filled by the CDNA5
// async global->LDS DMA engine; K-loop manually unrolled 2x so each
// buffer gets a compile-time identity (stable accumulator registers,
// no v_mov/v_nop churn around the WMMAs).
__global__ __launch_bounds__(256)
void k_gemm(const __bf16* __restrict__ A,
            const __bf16* __restrict__ Bw,
            const float*  __restrict__ bias,
            float*  __restrict__ hf,
            __bf16* __restrict__ hb,
            __bf16* __restrict__ zb,
            int mode)
{
    __shared__ __align__(16) unsigned short lsA[2][128 * 32];
    __shared__ __align__(16) unsigned short lsB[2][128 * 32];

    const int t    = threadIdx.x;
    const int lane = t & 31;
    const int wave = t >> 5;
    const int wm   = wave & 3;    // wave M index 0..3  (32 rows each)
    const int wn   = wave >> 2;   // wave N index 0..1  (64 cols each)
    const int half = lane >> 4;   // lane half 0/1
    const int l16  = lane & 15;

    const int m_blk = blockIdx.x * 128;
    const int n_blk = blockIdx.y * 128;

    v8f zero;
    #pragma unroll
    for (int e = 0; e < 8; ++e) zero[e] = 0.0f;
    v8f acc[2][4];
    #pragma unroll
    for (int i = 0; i < 2; ++i)
        #pragma unroll
        for (int j = 0; j < 4; ++j) acc[i][j] = zero;

    // staging: 128 rows x 32 cols bf16 per tile; thread t handles
    // row = t>>1, 16-element (32B) column group = t&1  (two b128 DMAs each)
    const int srow = t >> 1, scg = t & 1;
    const unsigned stoff = (unsigned)(srow * 64 + scg * 32);  // byte offset in tile

    const __bf16* aptr = A  + (size_t)(m_blk + srow) * HDIM + scg * 16;
    const __bf16* bptr = Bw + (size_t)(n_blk + srow) * HDIM + scg * 16;

    // LDS byte offsets (wave-relative, low 32 bits of the generic pointer)
    const unsigned ldsA[2] = { (unsigned)(size_t)(&lsA[0][0]) + stoff,
                               (unsigned)(size_t)(&lsA[1][0]) + stoff };
    const unsigned ldsB[2] = { (unsigned)(size_t)(&lsB[0][0]) + stoff,
                               (unsigned)(size_t)(&lsB[1][0]) + stoff };

    // prologue: DMA K-step 0 into buffer 0
    async_ld_b128(ldsA[0],      aptr);
    async_ld_b128(ldsA[0] + 16, aptr + 8);
    async_ld_b128(ldsB[0],      bptr);
    async_ld_b128(ldsB[0] + 16, bptr + 8);
    wait_asynccnt0();
    __syncthreads();

    const int NIT = HDIM / 32;   // 16 K-steps

    // one K-step, cur is a compile-time constant in each unrolled body
    auto kstep = [&](int it, int cur) {
        const bool more = (it + 1) < NIT;

        // kick off next tile's DMA into the other buffer; safe: that buffer
        // was last read in iteration it-1 and everyone passed that barrier.
        if (more) {
            const unsigned la = ldsA[cur ^ 1];
            const unsigned lb = ldsB[cur ^ 1];
            const __bf16* ga = aptr + (it + 1) * 32;
            const __bf16* gb = bptr + (it + 1) * 32;
            async_ld_b128(la,      ga);
            async_ld_b128(la + 16, ga + 8);
            async_ld_b128(lb,      gb);
            async_ld_b128(lb + 16, gb + 8);
        }

        // A fragments: 16x32 bf16 per ISA layout
        // lanes 0-15: K pairs 0..7 in v0..3, 16..23 in v4..7 ; lanes 16-31: +8
        FragU fa[2];
        #pragma unroll
        for (int i = 0; i < 2; ++i) {
            #pragma unroll
            for (int v = 0; v < 8; ++v) {
                int kk = (v < 4) ? (v * 2 + half * 8) : (16 + (v - 4) * 2 + half * 8);
                fa[i].u[v] = *reinterpret_cast<const unsigned int*>(
                    &lsA[cur][(wm * 32 + i * 16 + l16) * 32 + kk]);
            }
        }

        #pragma unroll
        for (int j = 0; j < 4; ++j) {
            // B fragment: 32x16; lanes 0-15: K 0..7,16..23 ; lanes 16-31: +8
            FragU fb;
            #pragma unroll
            for (int v = 0; v < 8; ++v) {
                int kk = (v < 4) ? (v * 2 + half * 8) : (16 + (v - 4) * 2 + half * 8);
                fb.u[v] = *reinterpret_cast<const unsigned int*>(
                    &lsB[cur][(wn * 64 + j * 16 + l16) * 32 + kk]);
            }
            acc[0][j] = __builtin_amdgcn_wmma_f32_16x16x32_bf16(
                false, fa[0].v, false, fb.v, (short)0, acc[0][j], false, false);
            acc[1][j] = __builtin_amdgcn_wmma_f32_16x16x32_bf16(
                false, fa[1].v, false, fb.v, (short)0, acc[1][j], false, false);
        }

        if (more) {
            // my DMAs done; barrier makes everyone's DMAs visible
            wait_asynccnt0();
            __syncthreads();
        }
    };

    // manual 2x unroll: buffer identity is compile-time in each body
    #pragma unroll 1
    for (int it2 = 0; it2 < NIT; it2 += 2) {
        kstep(it2 + 0, 0);
        kstep(it2 + 1, 1);
    }

    // epilogue: C layout -> m = half*8 + r, n = l16
    const int gn_base = n_blk + wn * 64;
    #pragma unroll
    for (int i = 0; i < 2; ++i) {
        const int gm_base = m_blk + wm * 32 + i * 16 + half * 8;
        #pragma unroll
        for (int j = 0; j < 4; ++j) {
            const int gn = gn_base + j * 16 + l16;
            const float bn = bias[gn];
            #pragma unroll
            for (int r = 0; r < 8; ++r) {
                const size_t o = (size_t)(gm_base + r) * HDIM + gn;
                float c = acc[i][j][r] + bn;
                if (mode == 0) {
                    float z = fmaxf(c, 0.0f);
                    zb[o] = (__bf16)z;
                } else {
                    float hn = hf[o] + c;
                    hf[o] = hn;
                    hb[o] = (__bf16)hn;
                }
            }
        }
    }
}

// ---------------- sigmoid heads + SIR recurrence ----------------
__global__ __launch_bounds__(256)
void k_heads_sir(const float* __restrict__ hf,
                 const float* __restrict__ Wb, const float* __restrict__ bb,
                 const float* __restrict__ Wg, const float* __restrict__ bg,
                 const float* __restrict__ x_state,  // (B, INLEN, NN, 3)
                 float* __restrict__ out)            // (B, OUTLEN, NN, 1)
{
    __shared__ float sWb[HDIM];
    __shared__ float sWg[HDIM];
    for (int i = threadIdx.x; i < HDIM; i += blockDim.x) {
        sWb[i] = Wb[i];
        sWg[i] = Wg[i];
    }
    __syncthreads();

    const int m = blockIdx.x * blockDim.x + threadIdx.x;   // < MROWS
    const int batch = m >> 11;
    const int node  = m & (NNODES - 1);

    const float* hr = hf + (size_t)m * HDIM;
    float db = bb[0], dg = bg[0];
    for (int k = 0; k < HDIM; ++k) {
        float hv = hr[k];
        db += hv * sWb[k];
        dg += hv * sWg[k];
    }
    const float bpar = 1.0f / (1.0f + expf(-db));
    const float g    = 1.0f / (1.0f + expf(-dg));

    #define XS(t, c) x_state[((size_t)(batch * INLEN + (t)) * NNODES + node) * 3 + (c)]

    const float Npop = XS(INLEN - 1, 2);

    float Iin[INLEN];
    float Ih_prev = XS(0, 1) / g;
    float I_Tm1 = 0.0f;
    #pragma unroll
    for (int t = 0; t < INLEN - 1; ++t) {
        float Ih_next = XS(t + 1, 1) / g;
        float iin = fmaxf(Ih_next - Ih_prev + XS(t, 1), 0.0f);
        Iin[t] = iin;
        I_Tm1 += expf(-g * (float)(INLEN - 1 - t)) * iin;
        Ih_prev = Ih_next;
    }

    const float R_Tm1 = Npop - XS(INLEN - 2, 0);
    const float S_Tm1 = Npop - I_Tm1 - R_Tm1;
    const float Iin0  = bpar * S_Tm1 * I_Tm1 / Npop;
    const float Rin0  = g * Ih_prev;          // g * I_hist[-1]
    Iin[INLEN - 1] = Iin0;

    float S = S_Tm1 - Iin0;
    float I = 0.0f;
    #pragma unroll
    for (int t = 0; t < INLEN; ++t)
        I += expf(-g * (float)(INLEN - t)) * Iin[t];
    float R = R_Tm1 + Rin0;
    (void)R;

    const float decay = expf(-g);
    for (int t = 0; t < OUTLEN; ++t) {
        float iin = bpar * S * I / Npop;
        S -= iin;
        float rin = g * I;
        I = decay * (I + iin);
        out[(size_t)(batch * OUTLEN + t) * NNODES + node] = rin;
    }
    #undef XS
}

// ---------------- host launch ----------------
extern "C" void kernel_launch(void* const* d_in, const int* in_sizes, int n_in,
                              void* d_out, int out_size, void* d_ws, size_t ws_size,
                              hipStream_t stream)
{
    const float* x_node   = (const float*)d_in[0];
    const float* x_state  = (const float*)d_in[1];
    const float* node_emb = (const float*)d_in[2];
    const float* W_ts     = (const float*)d_in[3];
    const float* b_ts     = (const float*)d_in[4];
    const float* enc_W1   = (const float*)d_in[5];
    const float* enc_b1   = (const float*)d_in[6];
    const float* enc_W2   = (const float*)d_in[7];
    const float* enc_b2   = (const float*)d_in[8];
    const float* Wb       = (const float*)d_in[9];
    const float* bb       = (const float*)d_in[10];
    const float* Wg       = (const float*)d_in[11];
    const float* bg       = (const float*)d_in[12];
    float* out = (float*)d_out;

    // workspace carve-up
    char* ws = (char*)d_ws;
    const size_t S_hf = (size_t)MROWS * HDIM * sizeof(float);   // 134 MB
    const size_t S_hb = (size_t)MROWS * HDIM * sizeof(__bf16);  //  67 MB
    const size_t S_w  = (size_t)NLAY * HDIM * HDIM * sizeof(__bf16);

    float*  hf  = (float*)ws;                       ws += S_hf;
    __bf16* hb  = (__bf16*)ws;                      ws += S_hb;
    __bf16* zb  = (__bf16*)ws;                      ws += S_hb;
    __bf16* W1b = (__bf16*)ws;                      ws += S_w;
    __bf16* W2b = (__bf16*)ws;                      ws += S_w;

    // 1) weights -> bf16
    k_convert_weights<<<(NLAY * HDIM * HDIM) / 256, 256, 0, stream>>>(
        enc_W1, enc_W2, W1b, W2b);

    // 2) h0 = [ts | node_emb]
    k_build_h0<<<((size_t)MROWS * HDIM) / 256, 256, 0, stream>>>(
        x_node, node_emb, W_ts, b_ts, hf, hb);

    // 3) encoder layers (bf16 WMMA GEMMs, fused epilogues)
    dim3 ggrid(MROWS / 128, HDIM / 128);
    for (int l = 0; l < NLAY; ++l) {
        k_gemm<<<ggrid, 256, 0, stream>>>(
            hb, W1b + (size_t)l * HDIM * HDIM, enc_b1 + l * HDIM,
            hf, hb, zb, /*mode=*/0);
        k_gemm<<<ggrid, 256, 0, stream>>>(
            zb, W2b + (size_t)l * HDIM * HDIM, enc_b2 + l * HDIM,
            hf, hb, zb, /*mode=*/1);
    }

    // 4) heads + SIR recurrence
    k_heads_sir<<<MROWS / 256, 256, 0, stream>>>(
        hf, Wb, bb, Wg, bg, x_state, out);
}
